// MPNN_36455682408490
// MI455X (gfx1250) — matrix operand order repaired and checked by reading, compiled, not run on hardware
//
#include <hip/hip_runtime.h>

// CDNA5 / gfx1250 WMMA types
typedef __attribute__((ext_vector_type(16))) __bf16 v16bf;
typedef __attribute__((ext_vector_type(8)))  float  v8f;

namespace {
constexpr int N_MOLS = 2048;
constexpr int APM    = 32;               // atoms per molecule
constexpr int NA     = N_MOLS * APM;     // 65536 atoms
constexpr int NB     = 2 * N_MOLS * 32;  // 131072 directed bonds
constexpr int AF     = 133;              // atom feature dim
constexpr int BF     = 147;              // bond feature dim
constexpr int H      = 512;              // hidden

constexpr int BM = 64;                   // block tile M
constexpr int BN = 128;                  // block tile N
constexpr int KS = 32;                   // WMMA K step (bf16 16x16x32)
constexpr int KP = KS + 8;               // padded LDS row stride (80B, 16B-aligned)

enum { MODE_PLAIN = 0, MODE_GATHER_SUB = 1 };
}

// 16B-granular views of bf16 data for vector LDS traffic.
union BF16x8  { __bf16 h[8];  uint4 u;    };
union BF16x16 { v16bf  v;     uint4 u[2]; __bf16 h[16]; };

// ---------------------------------------------------------------------------
// Tiled bf16 WMMA GEMM:  D = epilogue( A' @ Bw )
//   MODE_PLAIN:       A'[e,k] = A[e,k]                      (row stride lda)
//   MODE_GATHER_SUB:  A'[e,k] = A[idxS[e],k] - A2[idxR[e],k]  (fused D-MPNN
//                     neighbor-gather minus reverse-bond; saves a 512MB pass)
// Epilogue: +addend (ADD), +bias (BIAS), relu (RELU); STORE_RAW also writes
// the pre-relu value (keeps `inp` for the residual adds).
//
// 256 threads = 8 waves; wave grid 2(M) x 4(N); each wave: 2x2 WMMA tiles.
// Register-staged, LDS double-buffered pipeline, ONE barrier per K-tile:
//   issue next tile's global loads -> WMMA current tile from LDS (loads in
//   flight) -> cvt+store next tile to other LDS buffer -> barrier -> flip.
// A staged row-major [BM][KS]; B staged TRANSPOSED [BN][KS] so each lane's
// fragment is 32 contiguous bytes -> 2x ds_load_b128.
// Full K-tiles run branch-free; the (rare) remainder tile is peeled.
// ---------------------------------------------------------------------------
template <int MODE, bool AL4, bool RELU, bool ADD, bool BIAS, bool STORE_RAW>
__global__ __launch_bounds__(256)
void mpnn_gemm_wmma(const float* __restrict__ A, const float* __restrict__ A2,
                    const int* __restrict__ idxS, const int* __restrict__ idxR,
                    const float* __restrict__ Bw, const float* __restrict__ addend,
                    const float* __restrict__ bias,
                    float* __restrict__ D, float* __restrict__ Draw,
                    int M, int N, int K, int lda)
{
    __shared__ __bf16 As[2][BM][KP];   // 10 KB
    __shared__ __bf16 Bt[2][BN][KP];   // 20 KB (transposed: Bt[n][k])

    const int tid  = threadIdx.x;
    const int lane = tid & 31;
    const int wv   = tid >> 5;
    const int wm   = wv >> 2;       // 0..1
    const int wn   = wv & 3;        // 0..3
    const int bm0  = blockIdx.x * BM;
    const int bn0  = blockIdx.y * BN;
    const int half = lane >> 4;     // wave32 half select
    const int l15  = lane & 15;

    v8f acc[2][2] = {};

    // A staging: 4 threads per row, 8 consecutive k each
    const int a_row = tid >> 2;            // 0..63
    const int a_kc  = (tid & 3) * 8;       // 0,8,16,24
    const float* Arow;
    const float* Rrow = nullptr;
    if constexpr (MODE == MODE_GATHER_SUB) {
        Arow = A  + (size_t)idxS[bm0 + a_row] * lda;   // bond_src[e] -> a_msg row
        Rrow = A2 + (size_t)idxR[bm0 + a_row] * lda;   // b2revb[e]   -> msg row
    } else {
        Arow = A + (size_t)(bm0 + a_row) * lda;
    }
    // B staging (transposed): thread -> one n column, 16 k values
    const int b_n  = tid & 127;            // 0..127
    const int b_kh = (tid >> 7) * 16;      // 0 or 16

    float rA[8], rR[8], rB[16];

    // ---- pipeline stages -------------------------------------------------
    auto load_regs = [&](int k0) {            // global loads only, no waits
        if constexpr (MODE == MODE_GATHER_SUB) {
            const float4* ps = (const float4*)(Arow + k0 + a_kc);
            const float4* pr = (const float4*)(Rrow + k0 + a_kc);
            const float4 s0 = ps[0], s1 = ps[1];
            const float4 r0 = pr[0], r1 = pr[1];
            rA[0]=s0.x; rA[1]=s0.y; rA[2]=s0.z; rA[3]=s0.w;
            rA[4]=s1.x; rA[5]=s1.y; rA[6]=s1.z; rA[7]=s1.w;
            rR[0]=r0.x; rR[1]=r0.y; rR[2]=r0.z; rR[3]=r0.w;
            rR[4]=r1.x; rR[5]=r1.y; rR[6]=r1.z; rR[7]=r1.w;
        } else if constexpr (AL4) {
            const float4* p = (const float4*)(Arow + k0 + a_kc);
            const float4 x0 = p[0], x1 = p[1];
            rA[0]=x0.x; rA[1]=x0.y; rA[2]=x0.z; rA[3]=x0.w;
            rA[4]=x1.x; rA[5]=x1.y; rA[6]=x1.z; rA[7]=x1.w;
        } else {
            const float* p = Arow + k0 + a_kc;
#pragma unroll
            for (int j = 0; j < 8; ++j) rA[j] = p[j];
        }
#pragma unroll
        for (int i = 0; i < 16; ++i)
            rB[i] = Bw[(size_t)(k0 + b_kh + i) * N + (bn0 + b_n)];
    };

    auto store_lds = [&](int buf) {           // cvt fp32->bf16 + packed stores
        BF16x8 w;
#pragma unroll
        for (int j = 0; j < 8; ++j) {
            if constexpr (MODE == MODE_GATHER_SUB) w.h[j] = (__bf16)(rA[j] - rR[j]);
            else                                   w.h[j] = (__bf16)rA[j];
        }
        *(uint4*)&As[buf][a_row][a_kc] = w.u;          // 16B-aligned (80B rows)
        BF16x8 lo, hi;
#pragma unroll
        for (int i = 0; i < 8; ++i) { lo.h[i] = (__bf16)rB[i]; hi.h[i] = (__bf16)rB[8 + i]; }
        *(uint4*)&Bt[buf][b_n][b_kh]     = lo.u;
        *(uint4*)&Bt[buf][b_n][b_kh + 8] = hi.u;
    };

    auto compute = [&](int buf) {             // 8x ds_load_b128 + 4x v_wmma
        v16bf afr[2], bfr[2];
#pragma unroll
        for (int mi = 0; mi < 2; ++mi) {
            // A 16x32: lane->M=l15; elems 0..15 K=half*16+i
            const __bf16* ap = &As[buf][wm * 32 + mi * 16 + l15][0];
            BF16x16 f;
            f.u[0] = *(const uint4*)(ap + half * 16);
            f.u[1] = *(const uint4*)(ap + half * 16 + 8);
            afr[mi] = f.v;
        }
#pragma unroll
        for (int ni = 0; ni < 2; ++ni) {
            // B 32x16: lane->N=l15; lanes 0-15 K=0..15, lanes 16-31 K=16..31
            const __bf16* bp = &Bt[buf][wn * 32 + ni * 16 + l15][half * 16];
            BF16x16 f;
            f.u[0] = *(const uint4*)(bp);
            f.u[1] = *(const uint4*)(bp + 8);
            bfr[ni] = f.v;
        }
#pragma unroll
        for (int mi = 0; mi < 2; ++mi)
#pragma unroll
            for (int ni = 0; ni < 2; ++ni)
                acc[mi][ni] = __builtin_amdgcn_wmma_f32_16x16x32_bf16(
                    false, afr[mi], false, bfr[ni], (short)0, acc[mi][ni],
                    false, false);
    };

    // ---- main pipeline over full K-tiles (branch-free body) --------------
    const int nFull = K / KS;
    const int kFull = nFull * KS;
    int buf = 0;
    if (nFull > 0) {
        load_regs(0);
        store_lds(0);
        __syncthreads();
        for (int t = 0; t < nFull; ++t) {
            const bool more = (t + 1 < nFull);
            if (more) load_regs((t + 1) * KS);   // global loads in flight
            compute(buf);                        // WMMA from LDS[buf]
            if (more) store_lds(buf ^ 1);        // fill other buffer
            __syncthreads();
            buf ^= 1;
        }
    }

    // ---- peeled remainder tile (K=147/133 GEMMs only) --------------------
    if (kFull < K) {
        // clamped in-bounds loads (batched), then zero-select; no exec churn
        float raw[8];
#pragma unroll
        for (int j = 0; j < 8; ++j) {
            const int k  = kFull + a_kc + j;
            const int kc = (k < K) ? k : (K - 1);
            raw[j] = Arow[kc];
        }
#pragma unroll
        for (int j = 0; j < 8; ++j)
            rA[j] = (kFull + a_kc + j < K) ? raw[j] : 0.0f;
        if constexpr (MODE == MODE_GATHER_SUB) {
            float rawr[8];
#pragma unroll
            for (int j = 0; j < 8; ++j) {
                const int k  = kFull + a_kc + j;
                const int kc = (k < K) ? k : (K - 1);
                rawr[j] = Rrow[kc];
            }
#pragma unroll
            for (int j = 0; j < 8; ++j)
                rR[j] = (kFull + a_kc + j < K) ? rawr[j] : 0.0f;
        }
        float rawb[16];
#pragma unroll
        for (int i = 0; i < 16; ++i) {
            const int k  = kFull + b_kh + i;
            const int kc = (k < K) ? k : (K - 1);
            rawb[i] = Bw[(size_t)kc * N + (bn0 + b_n)];
        }
#pragma unroll
        for (int i = 0; i < 16; ++i)
            rB[i] = (kFull + b_kh + i < K) ? rawb[i] : 0.0f;
        store_lds(buf);
        __syncthreads();
        compute(buf);
    }

    // ---- epilogue: f32 C/D layout (VGPR r -> row half*8+r, col l15) ------
#pragma unroll
    for (int mi = 0; mi < 2; ++mi) {
        const int row0 = bm0 + wm * 32 + mi * 16 + half * 8;
#pragma unroll
        for (int ni = 0; ni < 2; ++ni) {
            const int col = bn0 + wn * 32 + ni * 16 + l15;
#pragma unroll
            for (int r = 0; r < 8; ++r) {
                const size_t off = (size_t)(row0 + r) * N + col;
                float v = acc[mi][ni][r];
                if constexpr (ADD)       v += addend[off];
                if constexpr (BIAS)      v += bias[col];
                if constexpr (STORE_RAW) Draw[off] = v;
                if constexpr (RELU)      v = fmaxf(v, 0.0f);
                D[off] = v;
            }
        }
    }
}

// ---------------------------------------------------------------------------
// Segment-sum of incoming bond messages per atom, atomic-free: each molecule
// owns 64 contiguous directed bonds whose dst atoms lie in its own 32 atoms.
// Block = (molecule, 128-column slab); thread privately owns one LDS column.
// ---------------------------------------------------------------------------
__global__ __launch_bounds__(128)
void mpnn_segsum(const float* __restrict__ msg, const int* __restrict__ bond_dst,
                 float* __restrict__ a_msg)
{
    __shared__ float acc[APM][128];        // 16 KB, conflict-free per-lane columns
    const int mol = blockIdx.x;
    const int hc  = blockIdx.y * 128 + threadIdx.x;
#pragma unroll
    for (int a = 0; a < APM; ++a) acc[a][threadIdx.x] = 0.0f;
    const int ebase = mol * 2 * APM;       // 64 directed bonds / molecule
    for (int e = 0; e < 2 * APM; ++e) {
        const int d = bond_dst[ebase + e] & (APM - 1);     // local atom id
        acc[d][threadIdx.x] += msg[(size_t)(ebase + e) * H + hc];
    }
#pragma unroll
    for (int a = 0; a < APM; ++a)
        a_msg[(size_t)(mol * APM + a) * H + hc] = acc[a][threadIdx.x];
}

// Mean pooling over the fixed 32 atoms of each molecule.
__global__ __launch_bounds__(128)
void mpnn_meanpool(const float* __restrict__ atom_h, float* __restrict__ mol_vec)
{
    const int mol = blockIdx.x;
    const int hc  = blockIdx.y * 128 + threadIdx.x;
    float s = 0.0f;
#pragma unroll
    for (int a = 0; a < APM; ++a) s += atom_h[(size_t)(mol * APM + a) * H + hc];
    mol_vec[(size_t)mol * H + hc] = s * (1.0f / APM);
}

// Final [M,512] @ [512,1] head: one wave32 per row, shuffle reduction.
__global__ __launch_bounds__(256)
void mpnn_head(const float* __restrict__ h, const float* __restrict__ outW,
               const float* __restrict__ outb, float* __restrict__ out)
{
    const int lane = threadIdx.x & 31;
    const int row  = blockIdx.x * 8 + (threadIdx.x >> 5);
    float s = 0.0f;
#pragma unroll
    for (int k = lane; k < H; k += 32)
        s += h[(size_t)row * H + k] * outW[k];
#pragma unroll
    for (int off = 16; off > 0; off >>= 1) s += __shfl_xor(s, off, 32);
    if (lane == 0) out[row] = s + outb[0];
}

// ---------------------------------------------------------------------------
extern "C" void kernel_launch(void* const* d_in, const int* in_sizes, int n_in,
                              void* d_out, int out_size, void* d_ws, size_t ws_size,
                              hipStream_t stream)
{
    const float* f_atoms = (const float*)d_in[0];
    const float* f_bonds = (const float*)d_in[1];
    const float* W_i     = (const float*)d_in[2];
    const float* W_h     = (const float*)d_in[3];
    const float* W_o     = (const float*)d_in[4];
    const float* c1W     = (const float*)d_in[5];
    const float* c1b     = (const float*)d_in[6];
    const float* c2W     = (const float*)d_in[7];
    const float* c2b     = (const float*)d_in[8];
    const float* c3W     = (const float*)d_in[9];
    const float* c3b     = (const float*)d_in[10];
    const float* outW    = (const float*)d_in[11];
    const float* outb    = (const float*)d_in[12];
    const int* bond_src  = (const int*)d_in[13];
    const int* bond_dst  = (const int*)d_in[14];
    const int* b2revb    = (const int*)d_in[15];
    // d_in[16] atom_mol_id: implicit in the contiguous 32-atom block layout.

    // Workspace layout (floats): inp + msgA + msgB + a_msg + 3 small = ~910 MB
    float* ws = (float*)d_ws;
    size_t o = 0;
    float* inp  = ws + o; o += (size_t)NB * H;       // pre-relu bond input
    float* msgA = ws + o; o += (size_t)NB * H;       // message ping
    float* msgB = ws + o; o += (size_t)NB * H;       // message pong
    float* amsg = ws + o; o += (size_t)NA * H;       // per-atom aggregate
    float* molv = ws + o; o += (size_t)N_MOLS * H;
    float* h1   = ws + o; o += (size_t)N_MOLS * H;
    float* h2   = ws + o; o += (size_t)N_MOLS * H;
    float* tmpP   = msgB;   // W_o partial: msgB is dead by then
    float* atom_h = inp;    // atom hidden: inp is dead by then

    const dim3 blk(256);
    const dim3 gB(NB / BM, H / BN);
    const dim3 gA(NA / BM, H / BN);
    const dim3 gM(N_MOLS / BM, H / BN);
    const dim3 segG(N_MOLS, H / 128), segB(128);

    // 1) inp = f_bonds @ W_i ; msgA = relu(inp)   (K=147, remainder peeled)
    mpnn_gemm_wmma<MODE_PLAIN, false, true, false, false, true>
        <<<gB, blk, 0, stream>>>(f_bonds, nullptr, nullptr, nullptr, W_i,
                                 nullptr, nullptr, msgA, inp, NB, H, BF, BF);

    // 2) depth iteration 1: msgB = relu(inp + (amsg[src]-msgA[rev]) @ W_h)
    mpnn_segsum<<<segG, segB, 0, stream>>>(msgA, bond_dst, amsg);
    mpnn_gemm_wmma<MODE_GATHER_SUB, true, true, true, false, false>
        <<<gB, blk, 0, stream>>>(amsg, msgA, bond_src, b2revb, W_h,
                                 inp, nullptr, msgB, nullptr, NB, H, H, H);

    // 3) depth iteration 2 (ping-pong back into msgA)
    mpnn_segsum<<<segG, segB, 0, stream>>>(msgB, bond_dst, amsg);
    mpnn_gemm_wmma<MODE_GATHER_SUB, true, true, true, false, false>
        <<<gB, blk, 0, stream>>>(amsg, msgB, bond_src, b2revb, W_h,
                                 inp, nullptr, msgA, nullptr, NB, H, H, H);

    // 4) final per-atom aggregation
    mpnn_segsum<<<segG, segB, 0, stream>>>(msgA, bond_dst, amsg);

    // 5) atom_h = relu([f_atoms | amsg] @ W_o), split along K (133 + 512)
    mpnn_gemm_wmma<MODE_PLAIN, false, false, false, false, false>
        <<<gA, blk, 0, stream>>>(f_atoms, nullptr, nullptr, nullptr, W_o,
                                 nullptr, nullptr, tmpP, nullptr, NA, H, AF, AF);
    mpnn_gemm_wmma<MODE_PLAIN, true, true, true, false, false>
        <<<gA, blk, 0, stream>>>(amsg, nullptr, nullptr, nullptr,
                                 W_o + (size_t)AF * H,
                                 tmpP, nullptr, atom_h, nullptr, NA, H, H, H);

    // 6) molecule mean pool (counts are exactly 32)
    mpnn_meanpool<<<dim3(N_MOLS, H / 128), segB, 0, stream>>>(atom_h, molv);

    // 7) classifier MLP (bias + relu fused in GEMM epilogue)
    mpnn_gemm_wmma<MODE_PLAIN, true, true, false, true, false>
        <<<gM, blk, 0, stream>>>(molv, nullptr, nullptr, nullptr, c1W,
                                 nullptr, c1b, h1, nullptr, N_MOLS, H, H, H);
    mpnn_gemm_wmma<MODE_PLAIN, true, true, false, true, false>
        <<<gM, blk, 0, stream>>>(h1, nullptr, nullptr, nullptr, c2W,
                                 nullptr, c2b, h2, nullptr, N_MOLS, H, H, H);
    mpnn_gemm_wmma<MODE_PLAIN, true, true, false, true, false>
        <<<gM, blk, 0, stream>>>(h2, nullptr, nullptr, nullptr, c3W,
                                 nullptr, c3b, h1, nullptr, N_MOLS, H, H, H);

    // 8) logits
    mpnn_head<<<dim3(N_MOLS / 8), dim3(256), 0, stream>>>(h1, outW, outb,
                                                          (float*)d_out);
    (void)in_sizes; (void)n_in; (void)out_size; (void)ws_size;
}
